// IMAGE_AI_MODEL_23733989278130
// MI455X (gfx1250) — matrix operand order repaired and checked by reading, not compiled, not run on hardware
//
#include <hip/hip_runtime.h>
#include <cstdint>
#include <cstddef>

#define HID   1024
#define VOC   50257
#define ENCD  2048
#define NSLOT 3

// ---- workspace layout (float-sized slots) ----
constexpr int WS_EMB    = 0;        // 1024  embedded token
constexpr int WS_ENC    = 1024;     // 3072  relu(enc @ fc_w.T + b)   [s*HID + h]
constexpr int WS_COMB   = 4096;     // 2048  [embedded ; attn_applied]
constexpr int WS_X      = 6144;     // 1024  relu(comb @ comb_w.T + b)
constexpr int WS_GATES  = 7168;     // 4096  LSTM preactivations
constexpr int WS_H1     = 11264;    // 1024  h1 (f32 staging)
constexpr int WS_H1PK   = 12288;    // 512 dwords: packed f16 pairs (h1[2p], h1[2p+1])
constexpr int WS_RED    = 12800;    // 2: [max, log(sum exp)]
constexpr int WS_LOGITS = 12816;    // 50257 logits (f32)

typedef _Float16 v16h __attribute__((ext_vector_type(16)));
typedef float    v8f  __attribute__((ext_vector_type(8)));
using h2 = decltype(__builtin_amdgcn_cvt_pkrtz(0.0f, 0.0f));

union A16 { v16h v; h2 p[8]; uint32_t u[8]; };

__device__ __forceinline__ float wred(float v) {
#pragma unroll
  for (int o = 16; o > 0; o >>= 1) v += __shfl_xor(v, o, 32);
  return v;
}
__device__ __forceinline__ float wredmax(float v) {
#pragma unroll
  for (int o = 16; o > 0; o >>= 1) v = fmaxf(v, __shfl_xor(v, o, 32));
  return v;
}
__device__ __forceinline__ float sigm(float x) { return 1.0f / (1.0f + expf(-x)); }

// ---- 0: embedding gather ----
__global__ void k_embed(const int* __restrict__ token, const float* __restrict__ emb,
                        float* __restrict__ ws) {
  int i = blockIdx.x * blockDim.x + threadIdx.x;
  if (i < HID) ws[WS_EMB + i] = emb[(size_t)token[0] * HID + i];
}

// ---- 1: enc = relu(encoder_outputs @ fc_w.T + fc_b), wave per output unit, 3 slots fused ----
__global__ void k_encproj(const float* __restrict__ enc_in, const float* __restrict__ fc_w,
                          const float* __restrict__ fc_b, float* __restrict__ ws) {
  int lane = threadIdx.x & 31;
  int h = (blockIdx.x * blockDim.x + threadIdx.x) >> 5;
  if (h >= HID) return;
  const float4* wr = (const float4*)(fc_w + (size_t)h * ENCD);
  const float4* e0 = (const float4*)(enc_in);
  const float4* e1 = (const float4*)(enc_in + ENCD);
  const float4* e2 = (const float4*)(enc_in + 2 * ENCD);
  float a0 = 0.f, a1 = 0.f, a2 = 0.f;
  for (int q = lane; q < ENCD / 4; q += 32) {
    float4 w = wr[q], x0 = e0[q], x1 = e1[q], x2 = e2[q];
    a0 += w.x * x0.x + w.y * x0.y + w.z * x0.z + w.w * x0.w;
    a1 += w.x * x1.x + w.y * x1.y + w.z * x1.z + w.w * x1.w;
    a2 += w.x * x2.x + w.y * x2.y + w.z * x2.z + w.w * x2.w;
  }
  a0 = wred(a0); a1 = wred(a1); a2 = wred(a2);
  if (lane == 0) {
    float b = fc_b[h];
    ws[WS_ENC + 0 * HID + h] = fmaxf(a0 + b, 0.f);
    ws[WS_ENC + 1 * HID + h] = fmaxf(a1 + b, 0.f);
    ws[WS_ENC + 2 * HID + h] = fmaxf(a2 + b, 0.f);
  }
}

// ---- 2: attention scores + softmax + context; build comb = [embedded ; context] ----
__global__ void k_attn(const float* __restrict__ hidden, const float* __restrict__ attn_w,
                       const float* __restrict__ attn_b, float* __restrict__ ws,
                       float* __restrict__ out) {
  __shared__ float sc[NSLOT];
  __shared__ float swt[NSLOT];
  int tid = threadIdx.x, lane = tid & 31, wv = tid >> 5;
  if (wv < NSLOT) {
    const float* wrow = attn_w + (size_t)wv * 2 * HID;
    float acc = 0.f;
    for (int j = lane; j < 2 * HID; j += 32) {
      float xin = (j < HID) ? ws[WS_EMB + j] : hidden[j - HID];
      acc += xin * wrow[j];
    }
    acc = wred(acc);
    if (lane == 0) sc[wv] = acc + attn_b[wv];
  }
  __syncthreads();
  if (tid == 0) {
    float m = fmaxf(sc[0], fmaxf(sc[1], sc[2]));
    float e0 = expf(sc[0] - m), e1 = expf(sc[1] - m), e2 = expf(sc[2] - m);
    float s = e0 + e1 + e2;
    swt[0] = e0 / s; swt[1] = e1 / s; swt[2] = e2 / s;
    out[VOC + 2 * HID + 0] = swt[0];
    out[VOC + 2 * HID + 1] = swt[1];
    out[VOC + 2 * HID + 2] = swt[2];
  }
  __syncthreads();
  for (int h = tid; h < HID; h += blockDim.x) {
    ws[WS_COMB + h] = ws[WS_EMB + h];
    ws[WS_COMB + HID + h] = swt[0] * ws[WS_ENC + h] +
                            swt[1] * ws[WS_ENC + HID + h] +
                            swt[2] * ws[WS_ENC + 2 * HID + h];
  }
}

// ---- 3: x = relu(comb @ comb_w.T + comb_b), wave per row ----
__global__ void k_comb(const float* __restrict__ comb_w, const float* __restrict__ comb_b,
                       float* __restrict__ ws) {
  int lane = threadIdx.x & 31;
  int r = (blockIdx.x * blockDim.x + threadIdx.x) >> 5;
  if (r >= HID) return;
  const float4* wr = (const float4*)(comb_w + (size_t)r * 2 * HID);
  const float4* xc = (const float4*)(ws + WS_COMB);
  float a = 0.f;
  for (int q = lane; q < (2 * HID) / 4; q += 32) {
    float4 w = wr[q], x = xc[q];
    a += w.x * x.x + w.y * x.y + w.z * x.z + w.w * x.w;
  }
  a = wred(a);
  if (lane == 0) ws[WS_X + r] = fmaxf(a + comb_b[r], 0.f);
}

// ---- 4: gates = x@w_ih.T + b_ih + h0@w_hh.T + b_hh, wave per row (both dots fused) ----
__global__ void k_gates(const float* __restrict__ w_ih, const float* __restrict__ w_hh,
                        const float* __restrict__ b_ih, const float* __restrict__ b_hh,
                        const float* __restrict__ hidden, float* __restrict__ ws) {
  int lane = threadIdx.x & 31;
  int r = (blockIdx.x * blockDim.x + threadIdx.x) >> 5;
  if (r >= 4 * HID) return;
  const float4* wi = (const float4*)(w_ih + (size_t)r * HID);
  const float4* wh = (const float4*)(w_hh + (size_t)r * HID);
  const float4* xv = (const float4*)(ws + WS_X);
  const float4* hv = (const float4*)(hidden);
  float a = 0.f;
  for (int q = lane; q < HID / 4; q += 32) {
    float4 A = wi[q], X = xv[q], B = wh[q], H = hv[q];
    a += A.x * X.x + A.y * X.y + A.z * X.z + A.w * X.w;
    a += B.x * H.x + B.y * H.y + B.z * H.z + B.w * H.w;
  }
  a = wred(a);
  if (lane == 0) ws[WS_GATES + r] = a + b_ih[r] + b_hh[r];
}

// ---- 5: LSTM pointwise; emit h1/c1 outputs + h1 packed f16 for WMMA B-operand ----
__global__ void k_lstm(const float* __restrict__ cell, float* __restrict__ ws,
                       float* __restrict__ out) {
  int h = threadIdx.x;
  float i = ws[WS_GATES + h];
  float f = ws[WS_GATES + HID + h];
  float g = ws[WS_GATES + 2 * HID + h];
  float o = ws[WS_GATES + 3 * HID + h];
  float c1 = sigm(f) * cell[h] + sigm(i) * tanhf(g);
  float h1 = sigm(o) * tanhf(c1);
  out[VOC + h] = h1;
  out[VOC + HID + h] = c1;
  ws[WS_H1 + h] = h1;
  __syncthreads();
  if (h < HID / 2) {
    union { h2 p; uint32_t u; } cv;
    cv.p = __builtin_amdgcn_cvt_pkrtz(ws[WS_H1 + 2 * h], ws[WS_H1 + 2 * h + 1]);
    ((uint32_t*)ws)[WS_H1PK + h] = cv.u;
  }
}

// ---- 6: vocab projection via V_WMMA_F32_16X16X32_F16; wave per 16 rows of out_w ----
// A (16x32, f16): lane<16 holds row M=lane, K = 0..7 & 16..23; lane>=16 row M=lane-16,
//                 K = 8..15 & 24..31 (documented 16-bit A layout). f32->f16 via cvt_pk_rtz.
// B (32x16, f16): every column carries the same h1 chunk (broadcast), so only the K->(vgpr,
//                 lane-half) mapping matters: lanes0-15 K=0..7 & 16..23, lanes16-31 K=8..15 & 24..31.
// D (16x16, f32): column 0 = dot products; lane 0 carries rows m0..m0+7, lane 16 rows +8..+15.
__global__ void k_vocab_wmma(const float* __restrict__ out_w, const float* __restrict__ out_b,
                             const uint32_t* __restrict__ h1pk, float* __restrict__ logits) {
  const int lane = threadIdx.x & 31;
  const int wave = (blockIdx.x * blockDim.x + threadIdx.x) >> 5;
  const int m0 = wave * 16;
  if (m0 >= ((VOC + 15) & ~15)) return;
  int row = m0 + (lane & 15);
  if (row >= VOC) row = VOC - 1;                       // clamp reads, guard stores
  const float* wrow = out_w + (size_t)row * HID;
  const int kb = (lane >> 4) << 3;                     // A: K sub-base 0 or 8
  const int bo = (lane >> 4) << 2;                     // B: pair-index base 0 or 4 (K 0/8)
  v8f c = {0.f, 0.f, 0.f, 0.f, 0.f, 0.f, 0.f, 0.f};
  for (int k0 = 0; k0 < HID; k0 += 32) {
    const float4* pa = (const float4*)(wrow + k0 + kb);
    float4 f0 = pa[0];
    float4 f1 = pa[1];
    const float4* pc = (const float4*)(wrow + k0 + kb + 16);
    float4 f2 = pc[0];
    float4 f3 = pc[1];
    A16 a;
    a.p[0] = __builtin_amdgcn_cvt_pkrtz(f0.x, f0.y);
    a.p[1] = __builtin_amdgcn_cvt_pkrtz(f0.z, f0.w);
    a.p[2] = __builtin_amdgcn_cvt_pkrtz(f1.x, f1.y);
    a.p[3] = __builtin_amdgcn_cvt_pkrtz(f1.z, f1.w);
    a.p[4] = __builtin_amdgcn_cvt_pkrtz(f2.x, f2.y);
    a.p[5] = __builtin_amdgcn_cvt_pkrtz(f2.z, f2.w);
    a.p[6] = __builtin_amdgcn_cvt_pkrtz(f3.x, f3.y);
    a.p[7] = __builtin_amdgcn_cvt_pkrtz(f3.z, f3.w);
    A16 b;
    const uint4* pb = (const uint4*)(h1pk + (k0 >> 1) + bo);
    uint4 b0 = pb[0], b1 = pb[2];
    b.u[0] = b0.x; b.u[1] = b0.y; b.u[2] = b0.z; b.u[3] = b0.w;
    b.u[4] = b1.x; b.u[5] = b1.y; b.u[6] = b1.z; b.u[7] = b1.w;
    c = __builtin_amdgcn_wmma_f32_16x16x32_f16(false, a.v, false, b.v,
                                               (short)0, c, false, false);
  }
  if ((lane & 15) == 0) {
    int base = m0 + ((lane >> 4) << 3);
#pragma unroll
    for (int i = 0; i < 8; ++i) {
      int r = base + i;
      if (r < VOC) logits[r] = c[i] + out_b[r];
    }
  }
}

// ---- 7: log-softmax reduction (max, log-sum-exp) over 50257 logits, one block ----
__global__ void k_lsred(float* __restrict__ ws) {
  __shared__ float redm[32];
  __shared__ float reds[32];
  int tid = threadIdx.x, lane = tid & 31, wv = tid >> 5;
  const float* lg = ws + WS_LOGITS;
  float m = -1e30f;
  for (int v = tid; v < VOC; v += blockDim.x) m = fmaxf(m, lg[v]);
  m = wredmax(m);
  if (lane == 0) redm[wv] = m;
  __syncthreads();
  if (wv == 0) {
    float t = (lane < (int)(blockDim.x >> 5)) ? redm[lane] : -1e30f;
    t = wredmax(t);
    if (lane == 0) redm[0] = t;
  }
  __syncthreads();
  m = redm[0];
  float s = 0.f;
  for (int v = tid; v < VOC; v += blockDim.x) s += expf(lg[v] - m);
  s = wred(s);
  if (lane == 0) reds[wv] = s;
  __syncthreads();
  if (tid == 0) {
    float t = 0.f;
    for (int i = 0; i < (int)(blockDim.x >> 5); ++i) t += reds[i];
    ws[WS_RED]     = m;
    ws[WS_RED + 1] = logf(t);
  }
}

// ---- 8: log_probs = logits - max - log(sum exp) ----
__global__ void k_lsout(const float* __restrict__ ws, float* __restrict__ out) {
  int v = blockIdx.x * blockDim.x + threadIdx.x;
  if (v < VOC) out[v] = ws[WS_LOGITS + v] - ws[WS_RED] - ws[WS_RED + 1];
}

extern "C" void kernel_launch(void* const* d_in, const int* in_sizes, int n_in,
                              void* d_out, int out_size, void* d_ws, size_t ws_size,
                              hipStream_t stream) {
  (void)in_sizes; (void)n_in; (void)out_size; (void)ws_size;
  const int*   token  = (const int*)  d_in[0];
  const float* hidden = (const float*)d_in[1];
  const float* cell   = (const float*)d_in[2];
  const float* enc_in = (const float*)d_in[3];
  const float* emb    = (const float*)d_in[4];
  const float* fc_w   = (const float*)d_in[5];
  const float* fc_b   = (const float*)d_in[6];
  const float* attn_w = (const float*)d_in[7];
  const float* attn_b = (const float*)d_in[8];
  const float* comb_w = (const float*)d_in[9];
  const float* comb_b = (const float*)d_in[10];
  const float* w_ih   = (const float*)d_in[11];
  const float* w_hh   = (const float*)d_in[12];
  const float* b_ih   = (const float*)d_in[13];
  const float* b_hh   = (const float*)d_in[14];
  const float* out_w  = (const float*)d_in[15];
  const float* out_b  = (const float*)d_in[16];
  float* out = (float*)d_out;
  float* ws  = (float*)d_ws;

  k_embed  <<<4,   256, 0, stream>>>(token, emb, ws);
  k_encproj<<<128, 256, 0, stream>>>(enc_in, fc_w, fc_b, ws);
  k_attn   <<<1,  1024, 0, stream>>>(hidden, attn_w, attn_b, ws, out);
  k_comb   <<<128, 256, 0, stream>>>(comb_w, comb_b, ws);
  k_gates  <<<512, 256, 0, stream>>>(w_ih, w_hh, b_ih, b_hh, hidden, ws);
  k_lstm   <<<1,  1024, 0, stream>>>(cell, ws, out);
  const int tiles = (VOC + 15) / 16;                       // 3142 wave-tiles
  k_vocab_wmma<<<(tiles + 7) / 8, 256, 0, stream>>>(out_w, out_b,
      (const uint32_t*)ws + WS_H1PK, ws + WS_LOGITS);
  k_lsred  <<<1,  1024, 0, stream>>>(ws);
  k_lsout  <<<(VOC + 255) / 256, 256, 0, stream>>>(ws, out);
}
